// VGAEconv_88751204204955
// MI455X (gfx1250) — matrix operand order repaired and checked by reading, not compiled, not run on hardware
//
#include <hip/hip_runtime.h>
#include <hip/hip_bf16.h>
#include <stdint.h>
#include <stddef.h>

// ---------------------------------------------------------------------------
// Types for CDNA5 WMMA (wave32): bf16 A/B operands (8 VGPRs of packed bf16),
// f32 C/D accumulators (8 VGPRs).
// ---------------------------------------------------------------------------
typedef __attribute__((ext_vector_type(16))) __bf16 v16bf;
typedef __attribute__((ext_vector_type(8)))  float  v8f;

union BF16x16 {
  v16bf v;
  unsigned short u[16];
  uint4 q[2];          // two 16B chunks -> ds_load_b128 / global_load_b128
};

__device__ __forceinline__ unsigned short f2bf(float f) {
  unsigned int x = __float_as_uint(f);
  x += 0x7FFFu + ((x >> 16) & 1u);          // round-to-nearest-even
  return (unsigned short)(x >> 16);
}
__device__ __forceinline__ unsigned int pack2bf(float a, float b) {
  return (unsigned int)f2bf(a) | ((unsigned int)f2bf(b) << 16);
}

// ---------------------------------------------------------------------------
// f32 -> bf16 packed conversion (2 elements / thread, b32 stores)
// ---------------------------------------------------------------------------
__global__ void to_bf16(const float* __restrict__ in,
                        unsigned int* __restrict__ out, size_t nPairs) {
  size_t i = (size_t)blockIdx.x * blockDim.x + threadIdx.x;
  if (i < nPairs) out[i] = pack2bf(in[2 * i], in[2 * i + 1]);
}

// ---------------------------------------------------------------------------
// Dense projection  Out[M x NOUT] = X[M x K] * W[K x NOUT]
//   X: bf16 row-major (pre-converted), W: f32 row-major, Out: f32.
// 8 waves/block; wave owns 16 rows, sweeps NOUT/16 column tiles.
// W is staged in LDS *pre-swizzled into the WMMA B register layout*:
//   per (n-tile, 32-k-block): 32 lanes x 16 contiguous bf16 (32 B/lane),
// so the B operand fill is two ds_load_b128. The A operand is two
// global_load_b128 straight from row-major bf16 (layout matches ISA 7.12.2).
// ---------------------------------------------------------------------------
template<int K, int NOUT>
__global__ void __launch_bounds__(256)
gemm_bf16_wmma(const unsigned short* __restrict__ Xb,
               const float* __restrict__ W,
               float* __restrict__ Out, int M) {
  constexpr int KS = 128;                   // K-stage (K is 128 or 256 here)
  constexpr int KB = KS / 32;               // 32-k blocks per stage
  constexpr int NT = NOUT / 16;             // 16-col tiles
  __shared__ unsigned int sW[KS * NOUT / 2];  // swizzled bf16 pairs (32/16 KB)

  const int lane = threadIdx.x & 31;
  const int wv   = threadIdx.x >> 5;
  const int lo   = lane & 15;
  const int hi   = lane >> 4;
  const int m0   = blockIdx.x * 128 + wv * 16;

  v8f acc[NT] = {};

  for (int ks = 0; ks < K; ks += KS) {
    // ---- stage W rows [ks, ks+KS) into LDS, swizzled to B-operand order ---
    // swizzled ushort index: ((t*KB + kb)*32 + lane)*16 + (2j + p)
    //   holds W[ks + kb*32 + 2(j&3) + p + 8*(lane>>4) + 16*(j>>2)][t*16 + (lane&15)]
    for (int pairIdx = threadIdx.x; pairIdx < KS * NOUT / 2; pairIdx += 256) {
      const int sidx    = pairIdx << 1;
      const int tileIdx = sidx >> 9;          // /(32 lanes * 16 elems)
      const int rem     = sidx & 511;
      const int lane_l  = rem >> 4;
      const int j       = (rem & 15) >> 1;
      const int t       = tileIdx / KB;
      const int kb      = tileIdx - t * KB;
      const int k  = kb * 32 + 2 * (j & 3) + 8 * (lane_l >> 4) + 16 * (j >> 2);
      const int n  = t * 16 + (lane_l & 15);
      const size_t g = (size_t)(ks + k) * NOUT + n;
      sW[pairIdx] = pack2bf(W[g], W[g + NOUT]);
    }
    __syncthreads();

    if (m0 < M) {                             // wave-uniform: EXEC all-1s
      for (int k0 = 0; k0 < KS; k0 += 32) {
        // A tile 16x32 bf16 from row-major bf16: VGPRs 0-3 = K+0..7,
        // VGPRs 4-7 = K+16..23, lane-half adds 8 (ISA 7.12.2).
        BF16x16 a;
        const unsigned short* xr =
            Xb + (size_t)(m0 + lo) * K + (ks + k0) + 8 * hi;
        a.q[0] = *(const uint4*)(xr);
        a.q[1] = *(const uint4*)(xr + 16);

        const int kb = k0 >> 5;
        #pragma unroll
        for (int t = 0; t < NT; ++t) {
          BF16x16 b;
          const uint4* bp =
              (const uint4*)&sW[(size_t)((t * KB + kb) * 32 + lane) * 8];
          b.q[0] = bp[0];
          b.q[1] = bp[1];
          acc[t] = __builtin_amdgcn_wmma_f32_16x16x32_bf16(
              false, a.v, false, b.v, (short)0, acc[t], false, false);
        }
      }
    }
    __syncthreads();
  }

  if (m0 < M) {
    // C/D layout: VGPR r -> row r (lanes 0-15) / row 8+r (lanes 16-31)
    #pragma unroll
    for (int t = 0; t < NT; ++t)
      #pragma unroll
      for (int r = 0; r < 8; ++r)
        Out[(size_t)(m0 + r + 8 * hi) * NOUT + t * 16 + lo] = acc[t][r];
  }
}

// ---------------------------------------------------------------------------
// Graph pipeline helper kernels
// ---------------------------------------------------------------------------
__global__ void fill_value(float* __restrict__ p, float v, size_t n) {
  size_t i = (size_t)blockIdx.x * blockDim.x + threadIdx.x;
  if (i < n) p[i] = v;
}

__global__ void deg_count(const long long* __restrict__ dst,
                          float* __restrict__ deg, long long E) {
  long long i = (long long)blockIdx.x * blockDim.x + threadIdx.x;
  if (i < E) atomicAdd(&deg[(size_t)dst[i]], 1.0f);
}

__global__ void to_rsqrt(float* __restrict__ p, size_t n) {
  size_t i = (size_t)blockIdx.x * blockDim.x + threadIdx.x;
  if (i < n) p[i] = rsqrtf(p[i]);           // deg >= 1 (self loop) always
}

__global__ void edge_norm(const long long* __restrict__ src,
                          const long long* __restrict__ dst,
                          const float* __restrict__ dis,
                          float* __restrict__ norm, long long E) {
  long long i = (long long)blockIdx.x * blockDim.x + threadIdx.x;
  if (i < E) norm[i] = dis[(size_t)src[i]] * dis[(size_t)dst[i]];
}

__global__ void bias_init(float* __restrict__ out, const float* __restrict__ b,
                          size_t total, int F) {
  size_t i = (size_t)blockIdx.x * blockDim.x + threadIdx.x;
  if (i < total) out[i] = b[i % F];
}

// scatter-add: agg[dst] += norm * msg[src], float4 chunks, atomics land in L2
template<int FEAT>
__global__ void scatter_edges(const long long* __restrict__ src,
                              const long long* __restrict__ dst,
                              const float* __restrict__ norm,
                              const float* __restrict__ msg,
                              float* __restrict__ agg, long long E) {
  constexpr int CH = FEAT / 4;
  size_t idx = (size_t)blockIdx.x * blockDim.x + threadIdx.x;
  size_t total = (size_t)E * CH;
  if (idx >= total) return;
  size_t e = idx / CH;
  int c = (int)(idx - e * CH) * 4;
  float w = norm[e];
  size_t s = (size_t)src[e], d = (size_t)dst[e];
  const float4 v = *(const float4*)(msg + s * FEAT + c);
  float* o = agg + d * FEAT + c;
  atomicAdd(o + 0, w * v.x);
  atomicAdd(o + 1, w * v.y);
  atomicAdd(o + 2, w * v.z);
  atomicAdd(o + 3, w * v.w);
}

// fused mu/sigma scatter (FEAT = 64): share edge + norm loads
__global__ void scatter_edges2(const long long* __restrict__ src,
                               const long long* __restrict__ dst,
                               const float* __restrict__ norm,
                               const float* __restrict__ mmu,
                               const float* __restrict__ mvar,
                               float* __restrict__ omu,
                               float* __restrict__ ovar, long long E) {
  constexpr int FEAT = 64, CH = FEAT / 4;
  size_t idx = (size_t)blockIdx.x * blockDim.x + threadIdx.x;
  size_t total = (size_t)E * CH;
  if (idx >= total) return;
  size_t e = idx / CH;
  int c = (int)(idx - e * CH) * 4;
  float w = norm[e];
  size_t s = (size_t)src[e], d = (size_t)dst[e];
  const float4 a = *(const float4*)(mmu + s * FEAT + c);
  const float4 b = *(const float4*)(mvar + s * FEAT + c);
  float* pm = omu  + d * FEAT + c;
  float* pv = ovar + d * FEAT + c;
  atomicAdd(pm + 0, w * a.x); atomicAdd(pm + 1, w * a.y);
  atomicAdd(pm + 2, w * a.z); atomicAdd(pm + 3, w * a.w);
  atomicAdd(pv + 0, w * b.x); atomicAdd(pv + 1, w * b.y);
  atomicAdd(pv + 2, w * b.z); atomicAdd(pv + 3, w * b.w);
}

// self-loop term + ReLU (layer 1); emits h directly as packed bf16 for GEMM2
template<int FEAT>
__global__ void epilogue_relu_bf16(const float* __restrict__ pre,
                                   const float* __restrict__ dis,
                                   const float* __restrict__ agg,
                                   unsigned int* __restrict__ hb,
                                   size_t nNodes) {
  size_t i = (size_t)blockIdx.x * blockDim.x + threadIdx.x;   // pair index
  size_t total = nNodes * (FEAT / 2);
  if (i >= total) return;
  size_t i2 = i * 2;
  size_t row = i2 / FEAT;
  float d = dis[row];
  float d2 = d * d;
  float v0 = fmaxf(agg[i2]     + d2 * pre[i2],     0.0f);
  float v1 = fmaxf(agg[i2 + 1] + d2 * pre[i2 + 1], 0.0f);
  hb[i] = pack2bf(v0, v1);
}

// self-loop term for mu and sigma outputs (FEAT = 64)
__global__ void epilogue_out(const float* __restrict__ mmu,
                             const float* __restrict__ mvar,
                             const float* __restrict__ dis,
                             float* __restrict__ omu,
                             float* __restrict__ ovar, size_t nNodes) {
  constexpr int FEAT = 64;
  size_t idx = (size_t)blockIdx.x * blockDim.x + threadIdx.x;
  size_t total = nNodes * FEAT;
  if (idx >= total) return;
  size_t row = idx / FEAT;
  float d = dis[row];
  float d2 = d * d;
  omu[idx]  += d2 * mmu[idx];
  ovar[idx] += d2 * mvar[idx];
}

// ---------------------------------------------------------------------------
// Launch
// ---------------------------------------------------------------------------
extern "C" void kernel_launch(void* const* d_in, const int* in_sizes, int n_in,
                              void* d_out, int out_size, void* d_ws, size_t ws_size,
                              hipStream_t stream) {
  const float*     x    = (const float*)d_in[0];
  const long long* ei   = (const long long*)d_in[1];  // int64 [2, E]
  const float*     W1   = (const float*)d_in[2];
  const float*     b1   = (const float*)d_in[3];
  const float*     Wmu  = (const float*)d_in[4];
  const float*     bmu  = (const float*)d_in[5];
  const float*     Wvar = (const float*)d_in[6];
  const float*     bvar = (const float*)d_in[7];

  const int F = 256, H = 128, D = 64;
  const long long N = in_sizes[0] / F;
  const long long E = (long long)in_sizes[1] / 2;
  const long long* src = ei;
  const long long* dst = ei + E;

  // workspace carve-up: dis[N] f32, norm[E] f32, xW1[N*H] f32, agg1[N*H] f32,
  // hb[N*H] bf16, then a 51.2MB region shared by xb (bf16 x, live only until
  // GEMM1) and mmu/mvar (live only after GEMM1 inputs are dead).  ~186 MB.
  char* wsb = (char*)d_ws;
  float* dis  = (float*)wsb;                 wsb += (size_t)N * sizeof(float);
  float* norm = (float*)wsb;                 wsb += (size_t)E * sizeof(float);
  float* xW1  = (float*)wsb;                 wsb += (size_t)N * H * sizeof(float);
  float* agg1 = (float*)wsb;                 wsb += (size_t)N * H * sizeof(float);
  unsigned short* hb = (unsigned short*)wsb; wsb += (size_t)N * H * sizeof(unsigned short);
  // shared region:
  unsigned short* xb = (unsigned short*)wsb;            // N*F bf16  (51.2 MB)
  float* mmu  = (float*)wsb;                            // N*D f32   (25.6 MB)
  float* mvar = mmu + (size_t)N * D;                    // N*D f32   (25.6 MB)

  float* omu  = (float*)d_out;               // mu    [N, 64]
  float* ovar = omu + (size_t)N * D;         // sigma [N, 64]

  const int T = 256;
  auto nb = [](size_t n) { return (unsigned)((n + 255) / 256); };
  const unsigned gemmBlocks = (unsigned)((N + 127) / 128);

  // --- normalization: deg (with self loop) -> dis -> per-edge norm ---
  fill_value<<<nb((size_t)N), T, 0, stream>>>(dis, 1.0f, (size_t)N);
  deg_count<<<nb((size_t)E), T, 0, stream>>>(dst, dis, E);
  to_rsqrt<<<nb((size_t)N), T, 0, stream>>>(dis, (size_t)N);
  edge_norm<<<nb((size_t)E), T, 0, stream>>>(src, dst, dis, norm, E);

  // --- layer 1: h = relu(segsum(norm * (xW1)[src]) + self + b1) ---
  to_bf16<<<nb((size_t)N * F / 2), T, 0, stream>>>(x, (unsigned int*)xb,
                                                   (size_t)N * F / 2);
  gemm_bf16_wmma<256, 128><<<gemmBlocks, 256, 0, stream>>>(xb, W1, xW1, (int)N);
  bias_init<<<nb((size_t)N * H), T, 0, stream>>>(agg1, b1, (size_t)N * H, H);
  scatter_edges<128><<<nb((size_t)E * 32), T, 0, stream>>>(src, dst, norm, xW1, agg1, E);
  epilogue_relu_bf16<128><<<nb((size_t)N * H / 2), T, 0, stream>>>(
      xW1, dis, agg1, (unsigned int*)hb, (size_t)N);

  // --- layer 2: mu / sigma heads (fused edge scatter); mmu/mvar alias xb ---
  gemm_bf16_wmma<128, 64><<<gemmBlocks, 256, 0, stream>>>(hb, Wmu,  mmu,  (int)N);
  gemm_bf16_wmma<128, 64><<<gemmBlocks, 256, 0, stream>>>(hb, Wvar, mvar, (int)N);
  bias_init<<<nb((size_t)N * D), T, 0, stream>>>(omu,  bmu,  (size_t)N * D, D);
  bias_init<<<nb((size_t)N * D), T, 0, stream>>>(ovar, bvar, (size_t)N * D, D);
  scatter_edges2<<<nb((size_t)E * 16), T, 0, stream>>>(src, dst, norm, mmu, mvar, omu, ovar, E);
  epilogue_out<<<nb((size_t)N * D), T, 0, stream>>>(mmu, mvar, dis, omu, ovar, (size_t)N);
}
